// SyntaxTransferEncoderDecoder_70635032150609
// MI455X (gfx1250) — matrix operand order repaired and checked by reading, not compiled, not run on hardware
//
#include <hip/hip_runtime.h>
#include <hip/hip_bf16.h>

// ---------------------------------------------------------------------------
// CDNA5 (gfx1250) implementation of the Tree-LSTM encoder / attention decoder.
// Core matmuls run through v_wmma_f32_16x16x32_bf16 (f32 accumulate).
// Algorithmic restructuring:
//   * emb @ {W_iou, W_f, W_ih}.T precomputed per vocab row -> gathers.
//   * leaf-level hcat/ccat are h0/c0 broadcasts -> two 1x1024 mat-vecs.
//   * per-level U GEMMs (the real FLOPs) are bf16 WMMA GEMMs.
//   * GEMM does a 16x64 tile per wave: A fragment reused over 4 WMMAs,
//     4 waves/block stacked on M share B fragments through the WGP cache.
//   * attention K (bf16) feeds the scores WMMA GEMM; ctx is a deterministic
//     split-K reduction (no FP atomics).
// ---------------------------------------------------------------------------

typedef __bf16 bf16;
typedef __attribute__((ext_vector_type(16))) __bf16 v16bf;
typedef __attribute__((ext_vector_type(8)))  __bf16 v8bf;
typedef __attribute__((ext_vector_type(8)))  float  v8f;

#define HD 256

__device__ __forceinline__ float sigf(float x) { return 1.0f / (1.0f + __expf(-x)); }

// ---------------------------------------------------------------------------
// WMMA GEMM: out[M,N] = A[M,K](bf16,row-major) @ W[N,K](bf16,row-major)^T
//            + bias0[N] + bias1[N]
// One wave -> 16(M) x 64(N) tile (4 accumulators, A fragment reused).
// Block: 4 waves stacked on M -> 64x64 block tile. K % 32 == 0.
// Ragged M/N handled with row clamping + masked stores (EXEC stays full).
// ---------------------------------------------------------------------------
template <bool BF16_OUT>
__global__ void gemm_bf16_wmma(const bf16* __restrict__ A, const bf16* __restrict__ W,
                               const float* __restrict__ bias0, const float* __restrict__ bias1,
                               void* __restrict__ out, int M, int N, int K) {
    const int wave  = threadIdx.x >> 5;       // uniform within wave
    const int lane  = threadIdx.x & 31;
    const int mtile = blockIdx.y * 4 + wave;
    const int m0    = mtile * 16;
    const int n0    = blockIdx.x * 64;
    const int half  = lane >> 4;              // 0: lanes 0-15, 1: lanes 16-31
    const int l16   = lane & 15;

    // A fragment (16x32, MxK): lanes<16 hold K {0..7,16..23}, lanes>=16 +8.
    int arow = m0 + l16; if (arow >= M) arow = M - 1;
    const bf16* aptr = A + (size_t)arow * K + half * 8;

    // 4 B sub-tiles (32x16, KxN) from W rows: lanes<16 K {0..7,16..23}, lanes>=16 +8.
    const bf16* wptr[4];
    float bv[4];
#pragma unroll
    for (int s = 0; s < 4; ++s) {
        const int nc = n0 + 16 * s + l16;
        int wrow = nc; if (wrow >= N) wrow = N - 1;
        wptr[s] = W + (size_t)wrow * K + half * 8;
        float b = 0.0f;
        if (nc < N) {
            if (bias0) b += bias0[nc];
            if (bias1) b += bias1[nc];
        }
        bv[s] = b;
    }

    v8f acc[4];
#pragma unroll
    for (int s = 0; s < 4; ++s)
#pragma unroll
        for (int i = 0; i < 8; ++i) acc[s][i] = bv[s];

    for (int k0 = 0; k0 < K; k0 += 32) {
        union { v16bf v; v8bf h[2]; } af;
        af.h[0] = *(const v8bf*)(aptr + k0);
        af.h[1] = *(const v8bf*)(aptr + k0 + 16);
        union { v16bf v; v8bf h[2]; } bf[4];
#pragma unroll
        for (int s = 0; s < 4; ++s) {
            bf[s].h[0] = *(const v8bf*)(wptr[s] + k0);
            bf[s].h[1] = *(const v8bf*)(wptr[s] + k0 + 16);
        }
#pragma unroll
        for (int s = 0; s < 4; ++s)
            acc[s] = __builtin_amdgcn_wmma_f32_16x16x32_bf16(
                false, af.v, false, bf[s].v, (short)0, acc[s], false, false);
    }

#pragma unroll
    for (int s = 0; s < 4; ++s) {
        const int ncol = n0 + 16 * s + l16;
        if (ncol < N) {
#pragma unroll
            for (int v = 0; v < 8; ++v) {
                const int mrow = m0 + v + half * 8;
                if (mrow < M) {
                    if (BF16_OUT)
                        ((bf16*)out)[(size_t)mrow * N + ncol] = (bf16)acc[s][v];
                    else
                        ((float*)out)[(size_t)mrow * N + ncol] = acc[s][v];
                }
            }
        }
    }
}

// f32 -> bf16 conversion
__global__ void cvt_bf16_kernel(const float* __restrict__ x, bf16* __restrict__ y, int n) {
    int i = blockIdx.x * blockDim.x + threadIdx.x;
    if (i < n) y[i] = (bf16)x[i];
}

// Leaf-level tree cell: hcat/ccat are h0/c0 broadcasts (precomputed u0_* vectors).
__global__ void leaf_gate_kernel(const int* __restrict__ src, int src_off,
                                 const float* __restrict__ proj_iou,  // [V,768] incl b_iou+b_uiou
                                 const float* __restrict__ proj_f,    // [V,256] incl b_wf
                                 const float* __restrict__ u0_iou,    // [768]  h0@U_iou.T
                                 const float* __restrict__ u0_f,      // [1024] h0@U_f.T + b_uf
                                 const float* __restrict__ c0,        // [1024]
                                 float* __restrict__ c_out, bf16* __restrict__ h_out) {
    const int n = blockIdx.x, j = threadIdx.x;
    const int id = src[src_off + n];
    const float* pi = proj_iou + (size_t)id * 768;
    const float i = sigf(pi[j]          + u0_iou[j]);
    const float o = sigf(pi[HD + j]     + u0_iou[HD + j]);
    const float u = tanhf(pi[2*HD + j]  + u0_iou[2*HD + j]);
    const float pf = proj_f[(size_t)id * HD + j];
    float c = i * u;
#pragma unroll
    for (int a = 0; a < 4; ++a)
        c += sigf(pf + u0_f[a * HD + j]) * c0[a * HD + j];
    const float h = o * tanhf(c);
    c_out[(size_t)n * HD + j] = c;
    h_out[(size_t)n * HD + j] = (bf16)h;
}

// Interior tree cell; acc_* hold hcat@U_*.T (acc_f includes b_uf).
__global__ void level_gate_kernel(const int* __restrict__ src, int src_off,
                                  const float* __restrict__ proj_iou,
                                  const float* __restrict__ proj_f,
                                  const float* __restrict__ acc_iou,  // [B,768]
                                  const float* __restrict__ acc_f,    // [B,1024]
                                  const float* __restrict__ c_child,  // [4B,256]
                                  float* __restrict__ c_out, bf16* __restrict__ h_out,
                                  int dup_root) {
    const int n = blockIdx.x, j = threadIdx.x;
    const int id = src[src_off + n];
    const float* pi = proj_iou + (size_t)id * 768;
    const float* ai = acc_iou + (size_t)n * 768;
    const float i = sigf(pi[j]         + ai[j]);
    const float o = sigf(pi[HD + j]    + ai[HD + j]);
    const float u = tanhf(pi[2*HD + j] + ai[2*HD + j]);
    const float pf = proj_f[(size_t)id * HD + j];
    const float* af = acc_f + (size_t)n * 1024;
    float c = i * u;
#pragma unroll
    for (int a = 0; a < 4; ++a)
        c += sigf(pf + af[a * HD + j]) * c_child[((size_t)(4 * n + a)) * HD + j];
    const float h = o * tanhf(c);
    c_out[(size_t)n * HD + j] = c;
    const bf16 hb = (bf16)h;
    h_out[(size_t)n * HD + j] = hb;
    if (dup_root) h_out[(size_t)(n + 1) * HD + j] = hb;  // enc_h duplicates the root
}

// Decoder LSTM gates. g_hh = h_last@W_hh.T + b_hh (from GEMM); proj_ih incl b_ih.
__global__ void dec_gate_kernel(const int* __restrict__ tgt, int tgt_off,
                                const float* __restrict__ proj_ih, // [V,1024]
                                const float* __restrict__ g_hh,    // [M,1024]
                                const float* __restrict__ c_last,  // row 0 broadcast
                                float* __restrict__ cn, bf16* __restrict__ hn_bf) {
    const int m = blockIdx.x, j = threadIdx.x;
    const int id = tgt[tgt_off + m];
    const float* p = proj_ih + (size_t)id * 1024;
    const float* g = g_hh + (size_t)m * 1024;
    const float gi = sigf(p[j]          + g[j]);
    const float gf = sigf(p[HD + j]     + g[HD + j]);
    const float gg = tanhf(p[2*HD + j]  + g[2*HD + j]);
    const float go = sigf(p[3*HD + j]   + g[3*HD + j]);
    const float c = gf * c_last[j] + gi * gg;
    cn[(size_t)m * HD + j] = c;
    hn_bf[(size_t)m * HD + j] = (bf16)(go * tanhf(c));
}

// Row softmax over n elements (in place), one block per row.
__global__ void softmax_row_kernel(float* __restrict__ s, int n) {
    float* row = s + (size_t)blockIdx.x * n;
    __shared__ float red[256];
    const int t = threadIdx.x;
    float mx = -3.0e38f;
    for (int k = t; k < n; k += 256) mx = fmaxf(mx, row[k]);
    red[t] = mx; __syncthreads();
    for (int o = 128; o > 0; o >>= 1) { if (t < o) red[t] = fmaxf(red[t], red[t + o]); __syncthreads(); }
    mx = red[0]; __syncthreads();
    float sum = 0.0f;
    for (int k = t; k < n; k += 256) { float e = __expf(row[k] - mx); row[k] = e; sum += e; }
    red[t] = sum; __syncthreads();
    for (int o = 128; o > 0; o >>= 1) { if (t < o) red[t] += red[t + o]; __syncthreads(); }
    const float inv = 1.0f / red[0];
    __syncthreads();
    for (int k = t; k < n; k += 256) row[k] *= inv;
}

// ctx = attn @ V, deterministic split-K: partials then reduce.
__global__ void ctx_partial_kernel(const float* __restrict__ attn, const float* __restrict__ V,
                                   float* __restrict__ part, int n, int nchunks) {
    const int j = threadIdx.x, ch = blockIdx.x, m = blockIdx.y;
    const int per = (n + nchunks - 1) / nchunks;
    const int k0 = ch * per;
    int k1 = k0 + per; if (k1 > n) k1 = n;
    const float* arow = attn + (size_t)m * n;
    float s = 0.0f;
    for (int k = k0; k < k1; ++k) s += arow[k] * V[(size_t)k * HD + j];
    part[((size_t)m * nchunks + ch) * HD + j] = s;
}
__global__ void ctx_reduce_kernel(const float* __restrict__ part, float* __restrict__ ctx, int nchunks) {
    const int j = threadIdx.x, m = blockIdx.x;
    float s = 0.0f;
    for (int ch = 0; ch < nchunks; ++ch) s += part[((size_t)m * nchunks + ch) * HD + j];
    ctx[(size_t)m * HD + j] = s;
}

// hc = [hn | ctx] in bf16
__global__ void concat_hc_kernel(const bf16* __restrict__ hn, const float* __restrict__ ctx,
                                 bf16* __restrict__ hc) {
    const int m = blockIdx.x, j = threadIdx.x;
    hc[(size_t)m * 512 + j]        = hn[(size_t)m * HD + j];
    hc[(size_t)m * 512 + HD + j]   = (bf16)ctx[(size_t)m * HD + j];
}

__global__ void rep4_bf_kernel(const float* __restrict__ hr, bf16* __restrict__ rep) {
    const int j = threadIdx.x;
    const bf16 b = (bf16)hr[j];
#pragma unroll
    for (int m = 0; m < 4; ++m) rep[m * HD + j] = b;
}

// 20 pre-output rows: rows 0..3 = h_r + pos[r]; rows 4.. = h_ch[p] + pos[a]
__global__ void build_outpre_kernel(const float* __restrict__ h_r, const float* __restrict__ h_ch,
                                    const float* __restrict__ pos, bf16* __restrict__ outpre) {
    const int r = blockIdx.x, j = threadIdx.x;
    float v;
    if (r < 4) v = h_r[j] + pos[r * HD + j];
    else { const int rr = r - 4; v = h_ch[(rr >> 2) * HD + j] + pos[(rr & 3) * HD + j]; }
    outpre[(size_t)r * HD + j] = (bf16)v;
}

// ---------------------------------------------------------------------------
static inline void gemm(const bf16* A, const bf16* W, const float* b0, const float* b1,
                        void* out, int M, int N, int K, bool bf16out, hipStream_t s) {
    dim3 grid((N + 63) / 64, (M + 63) / 64), block(128);
    if (bf16out) gemm_bf16_wmma<true ><<<grid, block, 0, s>>>(A, W, b0, b1, out, M, N, K);
    else         gemm_bf16_wmma<false><<<grid, block, 0, s>>>(A, W, b0, b1, out, M, N, K);
}
static inline void cvt(const float* x, bf16* y, int n, hipStream_t s) {
    cvt_bf16_kernel<<<(n + 255) / 256, 256, 0, s>>>(x, y, n);
}

extern "C" void kernel_launch(void* const* d_in, const int* in_sizes, int n_in,
                              void* d_out, int out_size, void* d_ws, size_t ws_size,
                              hipStream_t stream) {
    (void)in_sizes; (void)n_in; (void)out_size; (void)ws_size;
    const int*   src      = (const int*)  d_in[0];
    const int*   tgt      = (const int*)  d_in[1];
    const float* emb      = (const float*)d_in[2];
    const float* h0       = (const float*)d_in[3];
    const float* c0       = (const float*)d_in[4];
    const float* W_iou    = (const float*)d_in[5];
    const float* b_iou    = (const float*)d_in[6];
    const float* U_iou    = (const float*)d_in[7];
    const float* b_uiou   = (const float*)d_in[8];
    const float* W_f      = (const float*)d_in[9];
    const float* b_wf     = (const float*)d_in[10];
    const float* U_f      = (const float*)d_in[11];
    const float* b_uf     = (const float*)d_in[12];
    const float* W_ih     = (const float*)d_in[13];
    const float* b_ih     = (const float*)d_in[14];
    const float* W_hh     = (const float*)d_in[15];
    const float* b_hh     = (const float*)d_in[16];
    const float* Wq       = (const float*)d_in[17];
    const float* Wk       = (const float*)d_in[18];
    const float* Wv       = (const float*)d_in[19];
    const float* lin_W    = (const float*)d_in[20];
    const float* lin_b    = (const float*)d_in[21];
    const float* pos_emb  = (const float*)d_in[22];
    const float* W_out    = (const float*)d_in[23];
    const float* b_out    = (const float*)d_in[24];

    // tree geometry
    int p4[9]; p4[0] = 1; for (int i = 1; i < 9; ++i) p4[i] = p4[i - 1] * 4;
    int offs[10]; offs[0] = 0; for (int l = 0; l < 9; ++l) offs[l + 1] = offs[l] + p4[l];
    int enc_off[9]; enc_off[8] = 0;
    for (int l = 7; l >= 0; --l) enc_off[l] = enc_off[l + 1] + p4[l + 1];
    const int NNODES = enc_off[0] + 2;           // 87382 (root duplicated)
    const int NCH = 64;                          // ctx split-K chunks

    // ---- workspace bump allocator --------------------------------------
    char* base = (char*)d_ws;
    size_t off = 0;
    auto alloc = [&](size_t bytes) -> char* {
        char* p = base + off;
        off += (bytes + 255) & ~(size_t)255;
        return p;
    };
    bf16* emb_bf  = (bf16*)alloc(1000 * 256 * 2);
    bf16* h0_bf   = (bf16*)alloc(1024 * 2);
    bf16* Wiou_bf = (bf16*)alloc(768 * 256 * 2);
    bf16* Uiou_bf = (bf16*)alloc(768 * 1024 * 2);
    bf16* Wf_bf   = (bf16*)alloc(256 * 256 * 2);
    bf16* Uf_bf   = (bf16*)alloc(1024 * 1024 * 2);
    bf16* Wih_bf  = (bf16*)alloc(1024 * 256 * 2);
    bf16* Whh_bf  = (bf16*)alloc(1024 * 256 * 2);
    bf16* Wq_bf   = (bf16*)alloc(256 * 256 * 2);
    bf16* Wk_bf   = (bf16*)alloc(256 * 256 * 2);
    bf16* Wv_bf   = (bf16*)alloc(256 * 256 * 2);
    bf16* linW_bf = (bf16*)alloc(256 * 512 * 2);
    bf16* Wout_bf = (bf16*)alloc(1000 * 256 * 2);
    float* proj_iou = (float*)alloc((size_t)1000 * 768 * 4);
    float* proj_f   = (float*)alloc((size_t)1000 * 256 * 4);
    float* proj_ih  = (float*)alloc((size_t)1000 * 1024 * 4);
    float* u0_iou   = (float*)alloc(768 * 4);
    float* u0_f     = (float*)alloc(1024 * 4);
    bf16*  h_all_bf = (bf16*)alloc((size_t)NNODES * 256 * 2);
    float* cA       = (float*)alloc((size_t)65536 * 256 * 4);
    float* cB       = (float*)alloc((size_t)16384 * 256 * 4);
    // acc buffers (tree phase) overlaid with K/V (attention phase): disjoint in time
    char*  region   = alloc(134218752);
    float* acc_iou  = (float*)(region);
    float* acc_f    = (float*)(region + 50331648);
    bf16*  K_bf     = (bf16*) (region);
    float* V_f      = (float*)(region + 44739584);
    float* scores   = (float*)alloc((size_t)4 * NNODES * 4);
    float* part     = (float*)alloc((size_t)4 * NCH * 256 * 4);
    float* ctx      = (float*)alloc(4 * 256 * 4);
    float* g_hh     = (float*)alloc(4 * 1024 * 4);
    float* cn1      = (float*)alloc(256 * 4);
    float* cn2      = (float*)alloc(4 * 256 * 4);
    bf16*  hn1_bf   = (bf16*)alloc(256 * 2);
    bf16*  hn2_bf   = (bf16*)alloc(4 * 256 * 2);
    bf16*  q_bf     = (bf16*)alloc(4 * 256 * 2);
    bf16*  hc_bf    = (bf16*)alloc(4 * 512 * 2);
    float* h_r      = (float*)alloc(256 * 4);
    bf16*  rep_bf   = (bf16*)alloc(4 * 256 * 2);
    float* h_ch     = (float*)alloc(4 * 256 * 4);
    bf16*  outpre   = (bf16*)alloc(20 * 256 * 2);

    // ---- 1. bf16 conversions ------------------------------------------
    cvt(emb,   emb_bf,  1000 * 256, stream);
    cvt(h0,    h0_bf,   1024,       stream);
    cvt(W_iou, Wiou_bf, 768 * 256,  stream);
    cvt(U_iou, Uiou_bf, 768 * 1024, stream);
    cvt(W_f,   Wf_bf,   256 * 256,  stream);
    cvt(U_f,   Uf_bf,   1024 * 1024, stream);
    cvt(W_ih,  Wih_bf,  1024 * 256, stream);
    cvt(W_hh,  Whh_bf,  1024 * 256, stream);
    cvt(Wq,    Wq_bf,   256 * 256,  stream);
    cvt(Wk,    Wk_bf,   256 * 256,  stream);
    cvt(Wv,    Wv_bf,   256 * 256,  stream);
    cvt(lin_W, linW_bf, 256 * 512,  stream);
    cvt(W_out, Wout_bf, 1000 * 256, stream);

    // ---- 2. vocab-row projections + h0 projections ---------------------
    gemm(emb_bf, Wiou_bf, b_iou, b_uiou, proj_iou, 1000, 768, 256, false, stream);
    gemm(emb_bf, Wf_bf,   b_wf,  nullptr, proj_f,  1000, 256, 256, false, stream);
    gemm(emb_bf, Wih_bf,  b_ih,  nullptr, proj_ih, 1000, 1024, 256, false, stream);
    gemm(h0_bf,  Uiou_bf, nullptr, nullptr, u0_iou, 1, 768, 1024, false, stream);
    gemm(h0_bf,  Uf_bf,   b_uf,  nullptr, u0_f,    1, 1024, 1024, false, stream);

    // ---- 3. tree encoder ------------------------------------------------
    leaf_gate_kernel<<<65536, 256, 0, stream>>>(src, offs[8], proj_iou, proj_f,
                                                u0_iou, u0_f, c0,
                                                cA, h_all_bf /* level-8 at offset 0 */);
    float* cbufs[2] = { cA, cB };
    int cur = 0;                                  // children c live in cA
    for (int l = 7; l >= 0; --l) {
        const int B = p4[l];
        const bf16* Ah = h_all_bf + (size_t)enc_off[l + 1] * 256;   // hcat as B x 1024
        gemm(Ah, Uiou_bf, nullptr, nullptr, acc_iou, B, 768, 1024, false, stream);
        gemm(Ah, Uf_bf,   b_uf,   nullptr, acc_f,   B, 1024, 1024, false, stream);
        level_gate_kernel<<<B, 256, 0, stream>>>(src, offs[l], proj_iou, proj_f,
                                                 acc_iou, acc_f, cbufs[cur],
                                                 cbufs[cur ^ 1],
                                                 h_all_bf + (size_t)enc_off[l] * 256,
                                                 (l == 0) ? 1 : 0);
        cur ^= 1;
    }
    const float* c_root = cbufs[cur];             // row 0 holds root c

    // ---- 4. attention K/V over all encoder states ----------------------
    gemm(h_all_bf, Wk_bf, nullptr, nullptr, K_bf, NNODES, 256, 256, true,  stream);
    gemm(h_all_bf, Wv_bf, nullptr, nullptr, V_f,  NNODES, 256, 256, false, stream);

    const bf16* h_root_bf = h_all_bf + (size_t)(NNODES - 1) * 256;

    // ---- 5. decoder step 1 (root) --------------------------------------
    gemm(h_root_bf, Whh_bf, b_hh, nullptr, g_hh, 1, 1024, 256, false, stream);
    dec_gate_kernel<<<1, 256, 0, stream>>>(tgt, 0, proj_ih, g_hh, c_root, cn1, hn1_bf);
    gemm(hn1_bf, Wq_bf, nullptr, nullptr, q_bf, 1, 256, 256, true, stream);
    gemm(q_bf, K_bf, nullptr, nullptr, scores, 1, NNODES, 256, false, stream);
    softmax_row_kernel<<<1, 256, 0, stream>>>(scores, NNODES);
    ctx_partial_kernel<<<dim3(NCH, 1), 256, 0, stream>>>(scores, V_f, part, NNODES, NCH);
    ctx_reduce_kernel<<<1, 256, 0, stream>>>(part, ctx, NCH);
    concat_hc_kernel<<<1, 256, 0, stream>>>(hn1_bf, ctx, hc_bf);
    gemm(hc_bf, linW_bf, lin_b, nullptr, h_r, 1, 256, 512, false, stream);
    rep4_bf_kernel<<<1, 256, 0, stream>>>(h_r, rep_bf);

    // ---- 6. decoder step 2 (4 children) --------------------------------
    gemm(rep_bf, Whh_bf, b_hh, nullptr, g_hh, 4, 1024, 256, false, stream);
    dec_gate_kernel<<<4, 256, 0, stream>>>(tgt, 1, proj_ih, g_hh, cn1, cn2, hn2_bf);
    gemm(hn2_bf, Wq_bf, nullptr, nullptr, q_bf, 4, 256, 256, true, stream);
    gemm(q_bf, K_bf, nullptr, nullptr, scores, 4, NNODES, 256, false, stream);
    softmax_row_kernel<<<4, 256, 0, stream>>>(scores, NNODES);
    ctx_partial_kernel<<<dim3(NCH, 4), 256, 0, stream>>>(scores, V_f, part, NNODES, NCH);
    ctx_reduce_kernel<<<4, 256, 0, stream>>>(part, ctx, NCH);
    concat_hc_kernel<<<4, 256, 0, stream>>>(hn2_bf, ctx, hc_bf);
    gemm(hc_bf, linW_bf, lin_b, nullptr, h_ch, 4, 256, 512, false, stream);

    // ---- 7. output projection ------------------------------------------
    build_outpre_kernel<<<20, 256, 0, stream>>>(h_r, h_ch, pos_emb, outpre);
    gemm(outpre, Wout_bf, b_out, nullptr, d_out, 20, 1000, 256, false, stream);
}
